// ABNet_U_67250597920814
// MI455X (gfx1250) — matrix operand order repaired and checked by reading, not compiled, not run on hardware
//
#include <hip/hip_runtime.h>
#include <hip/hip_bf16.h>
#include <math.h>

// ---------- types ----------
typedef __attribute__((ext_vector_type(16))) __bf16 v16bf;
typedef __attribute__((ext_vector_type(8)))  float  v8f;
typedef __attribute__((ext_vector_type(4)))  unsigned int u32x4;
typedef int v4i_g __attribute__((vector_size(16)));   // matches builtin param type

union Frag { v16bf v; u32x4 q[2]; };

// CDNA5 async global->LDS path (guarded; falls back to sync staging)
#if defined(__has_builtin)
#  if __has_builtin(__builtin_amdgcn_global_load_async_to_lds_b128) && \
      __has_builtin(__builtin_amdgcn_s_wait_asynccnt)
#    define USE_ASYNC_LDS 1
#  endif
#endif
#ifndef USE_ASYNC_LDS
#  define USE_ASYNC_LDS 0
#endif

// Builtin signature: (AS1 v4i*, AS3 v4i*, i32 offset, i32 cpol). The immediate
// offset is added to BOTH the global and LDS addresses (ISA 10.7 async loads),
// so chunk #2 reuses the same address VGPRs with offset=16.
#define ASYNC_GPTR(p) ((__attribute__((address_space(1))) v4i_g*)(uintptr_t)(p))
#define ASYNC_LPTR(p) ((__attribute__((address_space(3))) v4i_g*)(uintptr_t)(p))

__device__ __forceinline__ unsigned short f2bf(float f) {
    unsigned int u = __builtin_bit_cast(unsigned int, f);
    u += 0x7fffu + ((u >> 16) & 1u);          // round-to-nearest-even
    return (unsigned short)(u >> 16);
}
__device__ __forceinline__ float bf2f(unsigned short h) {
    unsigned int u = ((unsigned int)h) << 16;
    return __builtin_bit_cast(float, u);
}

// ---------- weight fp32 -> bf16 ----------
__global__ void k_f32_to_bf16(const float* __restrict__ src,
                              unsigned short* __restrict__ dst, int n) {
    int i = blockIdx.x * blockDim.x + threadIdx.x;
    if (i < n) dst[i] = f2bf(src[i]);
}

// ---------- layer 1: h1 = relu(x @ W1^T + b1), K=4 (VALU path) ----------
__global__ __launch_bounds__(256) void k_layer1(
    const float* __restrict__ x, const float* __restrict__ W1,
    const float* __restrict__ b1, unsigned short* __restrict__ h1) {
    int i = blockIdx.x * blockDim.x + threadIdx.x;     // over M*2048
    int row = i >> 11;
    int n   = i & 2047;
    float acc = b1[n];
    #pragma unroll
    for (int k = 0; k < 4; ++k)
        acc = fmaf(x[row * 4 + k], W1[n * 4 + k], acc);
    h1[i] = f2bf(fmaxf(acc, 0.0f));
}

// ---------- main tiled bf16 WMMA GEMM: C = relu(A @ W^T + b) ----------
// A: M x K bf16 row-major, W: N x K bf16 row-major, C: M x N bf16
// 256 threads = 8 waves in a 2(M) x 4(N) grid; 128x128 tile; BK=32;
// 3-deep async LDS pipeline: tile it+2 in flight, wait asynccnt<=4 keeps the
// newest tile's loads outstanding across the barrier.
#define BM 128
#define BN 128
#define BK 32
#define LDT 40   // padded LDS row stride (halfs); 80B row stride, 16B aligned

__global__ __launch_bounds__(256) void k_gemm_bf16_relu(
    const unsigned short* __restrict__ A, const unsigned short* __restrict__ W,
    const float* __restrict__ bias, unsigned short* __restrict__ C,
    int M, int N, int K) {
#if USE_ASYNC_LDS
    __shared__ unsigned short sA[3][BM * LDT];
    __shared__ unsigned short sB[3][BN * LDT];
#else
    __shared__ unsigned short sA[2][BM * LDT];
    __shared__ unsigned short sB[2][BN * LDT];
#endif

    const int tid  = threadIdx.x;
    const int lane = tid & 31;
    const int wave = tid >> 5;
    const int wm = wave & 1;          // 2 waves along M
    const int wn = wave >> 1;         // 4 waves along N
    const int bm = blockIdx.y * BM;
    const int bn = blockIdx.x * BN;

    v8f acc[4][2];
    {
        v8f zero;
        #pragma unroll
        for (int e = 0; e < 8; ++e) zero[e] = 0.0f;
        #pragma unroll
        for (int i = 0; i < 4; ++i)
            #pragma unroll
            for (int j = 0; j < 2; ++j) acc[i][j] = zero;
    }

    // staging map: per tile each thread moves 16 halfs (2 x b128) per array
    const int lrow = tid >> 1;              // 0..127
    const int seg  = (tid & 1) * 16;        // 0 or 16 halfs
    const unsigned short* gA = A + (size_t)(bm + lrow) * K + seg;
    const unsigned short* gB = W + (size_t)(bn + lrow) * K + seg;
    const int ldsOff = lrow * LDT + seg;

    // fragment lane mapping (this code's contiguous k = 16*(lane/16) + s, 16-bit, wave32)
    const int arow  = lane & 15;            // A: M index
    const int koffA = (lane >> 4) * 16;     // A: lanes 16-31 take K 16..31
    const int bcol  = lane & 15;            // B: N index (row of W)
    const int koffB = (lane >> 4) * 16;     // B: lanes 16-31 take K 16..31

    const int nIter = K / BK;

#if USE_ASYNC_LDS
    // ---- 3-stage async pipeline: ASYNCcnt-tracked LDS loads, no VGPR round-trip
    {   // prologue: tiles 0 and 1 -> bufs 0 and 1
        __builtin_amdgcn_global_load_async_to_lds_b128(ASYNC_GPTR(gA),      ASYNC_LPTR(&sA[0][ldsOff]), 0,  0);
        __builtin_amdgcn_global_load_async_to_lds_b128(ASYNC_GPTR(gA),      ASYNC_LPTR(&sA[0][ldsOff]), 16, 0);
        __builtin_amdgcn_global_load_async_to_lds_b128(ASYNC_GPTR(gB),      ASYNC_LPTR(&sB[0][ldsOff]), 0,  0);
        __builtin_amdgcn_global_load_async_to_lds_b128(ASYNC_GPTR(gB),      ASYNC_LPTR(&sB[0][ldsOff]), 16, 0);
        __builtin_amdgcn_global_load_async_to_lds_b128(ASYNC_GPTR(gA + BK), ASYNC_LPTR(&sA[1][ldsOff]), 0,  0);
        __builtin_amdgcn_global_load_async_to_lds_b128(ASYNC_GPTR(gA + BK), ASYNC_LPTR(&sA[1][ldsOff]), 16, 0);
        __builtin_amdgcn_global_load_async_to_lds_b128(ASYNC_GPTR(gB + BK), ASYNC_LPTR(&sB[1][ldsOff]), 0,  0);
        __builtin_amdgcn_global_load_async_to_lds_b128(ASYNC_GPTR(gB + BK), ASYNC_LPTR(&sB[1][ldsOff]), 16, 0);
    }
    int cur = 0;      // buffer holding tile it
    int pre = 2;      // buffer that tile it+2 will be staged into
    for (int it = 0; it < nIter; ++it) {
        // async loads complete in order: <=4 outstanding means tile it is resident
        __builtin_amdgcn_s_wait_asynccnt(4);
        __syncthreads();
        if (it + 2 < nIter) {
            const int kk = (it + 2) * BK;
            __builtin_amdgcn_global_load_async_to_lds_b128(ASYNC_GPTR(gA + kk), ASYNC_LPTR(&sA[pre][ldsOff]), 0,  0);
            __builtin_amdgcn_global_load_async_to_lds_b128(ASYNC_GPTR(gA + kk), ASYNC_LPTR(&sA[pre][ldsOff]), 16, 0);
            __builtin_amdgcn_global_load_async_to_lds_b128(ASYNC_GPTR(gB + kk), ASYNC_LPTR(&sB[pre][ldsOff]), 0,  0);
            __builtin_amdgcn_global_load_async_to_lds_b128(ASYNC_GPTR(gB + kk), ASYNC_LPTR(&sB[pre][ldsOff]), 16, 0);
        }
        Frag a[4], b[2];
        #pragma unroll
        for (int mt = 0; mt < 4; ++mt) {
            const unsigned short* p = &sA[cur][(wm * 64 + mt * 16 + arow) * LDT + koffA];
            a[mt].q[0] = *(const u32x4*)(p);
            a[mt].q[1] = *(const u32x4*)(p + 8);
        }
        #pragma unroll
        for (int nt = 0; nt < 2; ++nt) {
            const unsigned short* p = &sB[cur][(wn * 32 + nt * 16 + bcol) * LDT + koffB];
            b[nt].q[0] = *(const u32x4*)(p);
            b[nt].q[1] = *(const u32x4*)(p + 8);
        }
        #pragma unroll
        for (int mt = 0; mt < 4; ++mt)
            #pragma unroll
            for (int nt = 0; nt < 2; ++nt)
                acc[mt][nt] = __builtin_amdgcn_wmma_f32_16x16x32_bf16(
                    false, a[mt].v, false, b[nt].v,
                    (short)0, acc[mt][nt], false, false);
        cur = (cur == 2) ? 0 : cur + 1;
        pre = (pre == 2) ? 0 : pre + 1;
    }
#else
    // ---- sync pipeline: global loads into regs overlap compute; ds_store after
    u32x4 ra0, ra1, rb0, rb1;
    {
        const u32x4* pa = (const u32x4*)(gA);
        const u32x4* pb = (const u32x4*)(gB);
        ra0 = pa[0]; ra1 = pa[1]; rb0 = pb[0]; rb1 = pb[1];
        u32x4* da = (u32x4*)&sA[0][ldsOff];
        u32x4* db = (u32x4*)&sB[0][ldsOff];
        da[0] = ra0; da[1] = ra1; db[0] = rb0; db[1] = rb1;
    }
    for (int it = 0; it < nIter; ++it) {
        const int cur = it & 1;
        __syncthreads();
        const bool more = (it + 1 < nIter);
        if (more) {
            const int kk = (it + 1) * BK;
            const u32x4* pa = (const u32x4*)(gA + kk);
            const u32x4* pb = (const u32x4*)(gB + kk);
            ra0 = pa[0]; ra1 = pa[1]; rb0 = pb[0]; rb1 = pb[1];
        }
        Frag a[4], b[2];
        #pragma unroll
        for (int mt = 0; mt < 4; ++mt) {
            const unsigned short* p = &sA[cur][(wm * 64 + mt * 16 + arow) * LDT + koffA];
            a[mt].q[0] = *(const u32x4*)(p);
            a[mt].q[1] = *(const u32x4*)(p + 8);
        }
        #pragma unroll
        for (int nt = 0; nt < 2; ++nt) {
            const unsigned short* p = &sB[cur][(wn * 32 + nt * 16 + bcol) * LDT + koffB];
            b[nt].q[0] = *(const u32x4*)(p);
            b[nt].q[1] = *(const u32x4*)(p + 8);
        }
        #pragma unroll
        for (int mt = 0; mt < 4; ++mt)
            #pragma unroll
            for (int nt = 0; nt < 2; ++nt)
                acc[mt][nt] = __builtin_amdgcn_wmma_f32_16x16x32_bf16(
                    false, a[mt].v, false, b[nt].v,
                    (short)0, acc[mt][nt], false, false);
        if (more) {
            const int nxt = cur ^ 1;
            u32x4* da = (u32x4*)&sA[nxt][ldsOff];
            u32x4* db = (u32x4*)&sB[nxt][ldsOff];
            da[0] = ra0; da[1] = ra1; db[0] = rb0; db[1] = rb1;
        }
    }
#endif

    // epilogue: C/D layout — lanes 0-15: (M=j, N=lane); lanes 16-31: (M=j+8, N=lane-16)
    const int cn  = lane & 15;
    const int mhi = (lane >> 4) * 8;
    #pragma unroll
    for (int nt = 0; nt < 2; ++nt) {
        const int gn = bn + wn * 32 + nt * 16 + cn;
        const float bv = bias[gn];
        #pragma unroll
        for (int mt = 0; mt < 4; ++mt) {
            #pragma unroll
            for (int j = 0; j < 8; ++j) {
                const int gm = bm + wm * 64 + mt * 16 + j + mhi;
                float f = fmaxf(acc[mt][nt][j] + bv, 0.0f);
                C[(size_t)gm * N + gn] = f2bf(f);
            }
        }
    }
}

// ---------- heads + CBF-QP epilogue: one wave (32 lanes) per row ----------
__global__ __launch_bounds__(256) void k_heads(
    const unsigned short* __restrict__ h41, const unsigned short* __restrict__ h42,
    const float* __restrict__ W51, const float* __restrict__ b51,
    const float* __restrict__ W52, const float* __restrict__ b52,
    const float* __restrict__ x,
    const float* __restrict__ mean, const float* __restrict__ stdv,
    const float* __restrict__ mean_label, const float* __restrict__ std_label,
    const float* __restrict__ wt, float* __restrict__ out) {
    const int lane = threadIdx.x & 31;
    const int wave = threadIdx.x >> 5;
    const int row  = blockIdx.x * 8 + wave;

    float a51[20], a52[11];
    #pragma unroll
    for (int o = 0; o < 20; ++o) a51[o] = 0.0f;
    #pragma unroll
    for (int o = 0; o < 11; ++o) a52[o] = 0.0f;

    const size_t base = (size_t)row * 1024;
    for (int k = lane; k < 1024; k += 32) {
        const float v1 = bf2f(h41[base + k]);
        const float v2 = bf2f(h42[base + k]);
        #pragma unroll
        for (int o = 0; o < 20; ++o) a51[o] = fmaf(v1, W51[o * 1024 + k], a51[o]);
        #pragma unroll
        for (int o = 0; o < 11; ++o) a52[o] = fmaf(v2, W52[o * 1024 + k], a52[o]);
    }
    // wave32 tree reduction
    #pragma unroll
    for (int off = 16; off > 0; off >>= 1) {
        #pragma unroll
        for (int o = 0; o < 20; ++o) a51[o] += __shfl_xor(a51[o], off, 32);
        #pragma unroll
        for (int o = 0; o < 11; ++o) a52[o] += __shfl_xor(a52[o], off, 32);
    }

    if (lane == 0) {
        float x51[20], x52[11];
        #pragma unroll
        for (int o = 0; o < 20; ++o) x51[o] = a51[o] + b51[o];
        #pragma unroll
        for (int o = 0; o < 11; ++o) {
            const float t = a52[o] + b52[o];
            x52[o] = 4.0f / (1.0f + expf(-t));
        }
        const float L1c = 3.0f, L2c = 3.0f, OX = 0.0f, OY = 7.0f, Rc = 4.0f;
        const float th1 = x[row * 4 + 0] * stdv[0] + mean[0];
        const float w1v = x[row * 4 + 1] * stdv[1] + mean[1];
        const float th2 = x[row * 4 + 2] * stdv[2] + mean[2];
        const float w2v = x[row * 4 + 3] * stdv[3] + mean[3];
        const float s1 = sinf(th1), c1 = cosf(th1);
        const float s2 = sinf(th2), c2 = cosf(th2);
        const float px = L1c * c1 + L2c * c2 - OX;
        const float py = L1c * s1 + L2c * s2 - OY;
        const float vx = -L1c * s1 * w1v - L2c * s2 * w2v;
        const float vy =  L1c * c1 * w1v + L2c * c2 * w2v;
        const float barrier = px * px + py * py - Rc * Rc;
        const float b_dot = 2.0f * (px * vx + py * vy);
        const float Lf2b = 2.0f * (vx * vx + vy * vy)
            + 2.0f * px * (-L1c * c1 * w1v * w1v - L2c * c2 * w2v * w2v)
            + 2.0f * py * (-L1c * s1 * w1v * w1v - L2c * s2 * w2v * w2v);
        const float g1 = 2.0f * px * (-L1c * s1) + 2.0f * py * (L1c * c1);
        const float g2 = 2.0f * px * (-L2c * s2) + 2.0f * py * (L2c * c2);
        const float Gdot = g1 * g1 + g2 * g2;
        // softmax(wt)
        float wmax = wt[0];
        #pragma unroll
        for (int h = 1; h < 10; ++h) wmax = fmaxf(wmax, wt[h]);
        float wv[10], wsum = 0.0f;
        #pragma unroll
        for (int h = 0; h < 10; ++h) { wv[h] = expf(wt[h] - wmax); wsum += wv[h]; }
        const float winv = 1.0f / wsum;
        const float alpha = x52[0];
        float rt0 = 0.0f, rt1 = 0.0f;
        #pragma unroll
        for (int h = 0; h < 10; ++h) {
            const float beta = x52[1 + h];
            const float hh = Lf2b + (alpha + beta) * b_dot + alpha * beta * barrier;
            const float r0 = x51[2 * h], r1 = x51[2 * h + 1];
            const float viol = r0 * g1 + r1 * g2 - hh;     // u_unc·G - h
            const float lam = fmaxf(viol / Gdot, 0.0f);
            const float wgt = wv[h] * winv;
            rt0 += wgt * (-r0 + lam * g1);
            rt1 += wgt * (-r1 + lam * g2);
        }
        out[row * 2 + 0] = (rt0 - mean_label[0]) / std_label[0];
        out[row * 2 + 1] = (rt1 - mean_label[1]) / std_label[1];
    }
}

// ---------- host ----------
extern "C" void kernel_launch(void* const* d_in, const int* in_sizes, int n_in,
                              void* d_out, int out_size, void* d_ws, size_t ws_size,
                              hipStream_t stream) {
    (void)in_sizes; (void)n_in; (void)out_size; (void)ws_size;
    const float* x          = (const float*)d_in[0];
    const float* mean       = (const float*)d_in[2];
    const float* stdv       = (const float*)d_in[3];
    const float* mean_label = (const float*)d_in[4];
    const float* std_label  = (const float*)d_in[5];
    const float* W1  = (const float*)d_in[6];  const float* b1  = (const float*)d_in[7];
    const float* W2  = (const float*)d_in[8];  const float* b2  = (const float*)d_in[9];
    const float* W31 = (const float*)d_in[10]; const float* b31 = (const float*)d_in[11];
    const float* W32 = (const float*)d_in[12]; const float* b32 = (const float*)d_in[13];
    const float* W41 = (const float*)d_in[14]; const float* b41 = (const float*)d_in[15];
    const float* W42 = (const float*)d_in[16]; const float* b42 = (const float*)d_in[17];
    const float* W51 = (const float*)d_in[18]; const float* b51 = (const float*)d_in[19];
    const float* W52 = (const float*)d_in[20]; const float* b52 = (const float*)d_in[21];
    const float* wt  = (const float*)d_in[22];
    float* out = (float*)d_out;

    const int M = 16384;
    char* ws = (char*)d_ws;
    const size_t MB = 1024u * 1024u;
    unsigned short* W2b  = (unsigned short*)(ws + 0);
    unsigned short* W31b = (unsigned short*)(ws + 8 * MB);
    unsigned short* W32b = (unsigned short*)(ws + 12 * MB);
    unsigned short* W41b = (unsigned short*)(ws + 16 * MB);
    unsigned short* W42b = (unsigned short*)(ws + 18 * MB);
    unsigned short* h1   = (unsigned short*)(ws + 20 * MB);   // 64 MB
    unsigned short* h2   = (unsigned short*)(ws + 84 * MB);   // 64 MB
    unsigned short* h31  = (unsigned short*)(ws + 148 * MB);  // 32 MB
    unsigned short* h32  = (unsigned short*)(ws + 180 * MB);  // 32 MB
    unsigned short* h41  = (unsigned short*)(ws + 20 * MB);   // reuse h1 (dead)
    unsigned short* h42  = (unsigned short*)(ws + 52 * MB);   // reuse h1 upper half

    // weight conversion fp32 -> bf16 (L2-resident thereafter)
    k_f32_to_bf16<<<(2048 * 2048 + 255) / 256, 256, 0, stream>>>(W2,  W2b,  2048 * 2048);
    k_f32_to_bf16<<<(1024 * 2048 + 255) / 256, 256, 0, stream>>>(W31, W31b, 1024 * 2048);
    k_f32_to_bf16<<<(1024 * 2048 + 255) / 256, 256, 0, stream>>>(W32, W32b, 1024 * 2048);
    k_f32_to_bf16<<<(1024 * 1024 + 255) / 256, 256, 0, stream>>>(W41, W41b, 1024 * 1024);
    k_f32_to_bf16<<<(1024 * 1024 + 255) / 256, 256, 0, stream>>>(W42, W42b, 1024 * 1024);

    k_layer1<<<M * 2048 / 256, 256, 0, stream>>>(x, W1, b1, h1);

    dim3 blk(256);
    k_gemm_bf16_relu<<<dim3(2048 / BN, M / BM), blk, 0, stream>>>(h1,  W2b,  b2,  h2,  M, 2048, 2048);
    k_gemm_bf16_relu<<<dim3(1024 / BN, M / BM), blk, 0, stream>>>(h2,  W31b, b31, h31, M, 1024, 2048);
    k_gemm_bf16_relu<<<dim3(1024 / BN, M / BM), blk, 0, stream>>>(h2,  W32b, b32, h32, M, 1024, 2048);
    k_gemm_bf16_relu<<<dim3(1024 / BN, M / BM), blk, 0, stream>>>(h31, W41b, b41, h41, M, 1024, 1024);
    k_gemm_bf16_relu<<<dim3(1024 / BN, M / BM), blk, 0, stream>>>(h32, W42b, b42, h42, M, 1024, 1024);

    k_heads<<<M / 8, 256, 0, stream>>>(h41, h42, W51, b51, W52, b52, x,
                                       mean, stdv, mean_label, std_label, wt, out);
}
